// Attention_82343112999676
// MI455X (gfx1250) — matrix operand order repaired and checked by reading, not compiled, not run on hardware
//
#include <hip/hip_runtime.h>
#include <hip/hip_bf16.h>

// ---------------------------------------------------------------------------
// Types for CDNA5 WMMA (wave32)
// ---------------------------------------------------------------------------
typedef __bf16 bf16_t;
typedef __attribute__((ext_vector_type(16))) __bf16 bf16x16;
typedef __attribute__((ext_vector_type(8)))  __bf16 bf16x8;
typedef __attribute__((ext_vector_type(8)))  float  floatx8;

static constexpr int BB  = 4;
static constexpr int SS  = 1024;
static constexpr int DIM = 4096;
static constexpr int NH  = 32;
static constexpr int HD  = 128;
static constexpr int RR  = 16;
static constexpr int AL  = 10;
static constexpr long MROWS = (long)BB * SS;          // 4096
static constexpr float LOG2E = 1.4426950408889634f;
static constexpr float SCALE = 0.08838834764831845f;  // 1/sqrt(128)

__device__ __forceinline__ floatx8 wmma_bf16(bf16x16 a, bf16x16 b, floatx8 c) {
  return __builtin_amdgcn_wmma_f32_16x16x32_bf16(false, a, false, b, (short)0, c,
                                                 false, false);
}

// A-fragment (16x32, MxK): lane<16 holds row M=lane, K[0..15];
// lane>=16 holds row M=lane-16, K[16..31]. p -> (row0, k0), ld elems.
__device__ __forceinline__ bf16x16 frag_a(const bf16_t* p, int ld, int lane) {
  int m = lane & 15, h = lane >> 4;
  const bf16_t* r = p + (long)m * ld + h * 16;
  bf16x8 lo = *(const bf16x8*)(r);
  bf16x8 hi = *(const bf16x8*)(r + 8);
  return __builtin_shufflevector(lo, hi, 0, 1, 2, 3, 4, 5, 6, 7,
                                 8, 9, 10, 11, 12, 13, 14, 15);
}

// B-fragment (32x16, KxN) where memory holds rows of the *N* dimension with K
// contiguous (i.e. W[n][k]): lane<16 -> col n, K[0..15]; lane>=16 -> K[16..31].
__device__ __forceinline__ bf16x16 frag_b(const bf16_t* p, int ld, int lane) {
  int n = lane & 15, h = lane >> 4;
  return *(const bf16x16*)(p + (long)n * ld + h * 16);
}

// gfx1250 async global->LDS copy (16B) + waits
__device__ __forceinline__ void async_b128(unsigned long long gbase, int byteOff,
                                           void* ldsPtr) {
  unsigned l = (unsigned)(unsigned long long)ldsPtr;  // low 32 bits == LDS addr
  asm volatile("global_load_async_to_lds_b128 %0, %1, %2"
               :: "v"(l), "v"(byteOff), "s"(gbase) : "memory");
}
__device__ __forceinline__ void wait_async() {
  asm volatile("s_wait_asynccnt 0" ::: "memory");
}
__device__ __forceinline__ void wait_ds() {
  asm volatile("s_wait_dscnt 0" ::: "memory");
}

// ---------------------------------------------------------------------------
// 1) fp32 -> bf16 conversion
// ---------------------------------------------------------------------------
__global__ void cvt_bf16_kernel(const float* __restrict__ in,
                                bf16_t* __restrict__ out, int n) {
  int i = blockIdx.x * 256 + threadIdx.x;
  if (i < n) out[i] = (bf16_t)in[i];
}

// ---------------------------------------------------------------------------
// 2) W_eff = W + l2 @ l1  (rank-16 fold of LoRA into dense weight), bf16 out
// ---------------------------------------------------------------------------
__global__ void merge_weight_kernel(const float* __restrict__ w,
                                    const float* __restrict__ l2,
                                    const float* __restrict__ l1,
                                    bf16_t* __restrict__ out) {
  int idx = blockIdx.x * 256 + threadIdx.x;   // over DIM*DIM (16M)
  int n = idx >> 12;                          // output row (o)
  int k = idx & (DIM - 1);                    // input col (d)
  float s = w[(long)idx];
  #pragma unroll
  for (int r = 0; r < RR; ++r) s += l2[n * RR + r] * l1[(long)r * DIM + k];
  out[(long)idx] = (bf16_t)s;
}

// ---------------------------------------------------------------------------
// 3) Tiled bf16 GEMM: C[M,N] = A[M,K] * W[N,K]^T.
//    BM=BN=128, BK=32, 256 threads = 8 waves (2x4), each wave 64x32.
//    Double-buffered LDS; async copy of tile i+1 overlaps WMMA on tile i.
// ---------------------------------------------------------------------------
__global__ __launch_bounds__(256)
void gemm_bf16_kernel(const bf16_t* __restrict__ A, const bf16_t* __restrict__ W,
                      void* __restrict__ C, int M, int N, int K, int out_f32) {
  __shared__ __align__(128) bf16_t At[2][128 * 32];
  __shared__ __align__(128) bf16_t Bt[2][128 * 32];
  const int tileM = blockIdx.y * 128, tileN = blockIdx.x * 128;
  const int t = threadIdx.x, lane = t & 31, w = t >> 5;
  const int wm = (w >> 2) * 64;   // 0 / 64
  const int wn = (w & 3) * 32;    // 0 / 32 / 64 / 96
  const unsigned long long aB = (unsigned long long)A;
  const unsigned long long wB = (unsigned long long)W;

  auto stage = [&](int buf, int k0) {
    #pragma unroll
    for (int itc = 0; itc < 2; ++itc) {
      int c = t + itc * 256;                 // 0..511, uniform trip count
      int row = c >> 2, seg = (c & 3) * 8;   // 4 x 8 bf16 segments per row
      async_b128(aB, (int)((((long)(tileM + row)) * K + k0 + seg) * 2),
                 &At[buf][row * 32 + seg]);
      async_b128(wB, (int)((((long)(tileN + row)) * K + k0 + seg) * 2),
                 &Bt[buf][row * 32 + seg]);
    }
  };

  floatx8 acc[4][2];
  #pragma unroll
  for (int mi = 0; mi < 4; ++mi)
    #pragma unroll
    for (int ni = 0; ni < 2; ++ni) acc[mi][ni] = (floatx8)(0.f);

  const int nIter = K >> 5;
  stage(0, 0);
  for (int itk = 0; itk < nIter; ++itk) {
    const int cur = itk & 1;
    wait_async();          // tile `itk` resident in LDS for this wave
    __syncthreads();       // ... for all waves; prior reads of buf `cur^1` done
    if (itk + 1 < nIter) stage(cur ^ 1, (itk + 1) << 5);   // overlap next copy
    #pragma unroll
    for (int mi = 0; mi < 4; ++mi) {
      bf16x16 af = frag_a(&At[cur][(wm + mi * 16) * 32], 32, lane);
      #pragma unroll
      for (int ni = 0; ni < 2; ++ni) {
        bf16x16 bf = frag_b(&Bt[cur][(wn + ni * 16) * 32], 32, lane);
        acc[mi][ni] = wmma_bf16(af, bf, acc[mi][ni]);
      }
    }
  }
  // epilogue: C layout -> VGPR i holds rows (i, i+8) split by lane half
  const int col = lane & 15, half = lane >> 4;
  #pragma unroll
  for (int mi = 0; mi < 4; ++mi)
    #pragma unroll
    for (int ni = 0; ni < 2; ++ni)
      #pragma unroll
      for (int i = 0; i < 8; ++i) {
        long gr = tileM + wm + mi * 16 + half * 8 + i;
        long gc = tileN + wn + ni * 16 + col;
        float v = acc[mi][ni][i];
        if (out_f32) ((float*)C)[gr * N + gc] = v;
        else         ((bf16_t*)C)[gr * N + gc] = (bf16_t)v;
      }
}

// ---------------------------------------------------------------------------
// 4) RoPE in place on bf16 [B,S,H,HD]
// ---------------------------------------------------------------------------
__global__ void rope_kernel(bf16_t* __restrict__ tq,
                            const float* __restrict__ cs,
                            const float* __restrict__ sn) {
  int idx = blockIdx.x * 256 + threadIdx.x;   // B*S*H*(HD/2) = 8388608
  int j = idx & 63;
  int h = (idx >> 6) & 31;
  int s = (idx >> 11) & 1023;
  int b = idx >> 21;
  long base = ((((long)b * SS + s) * NH + h) * HD) + 2 * j;
  float r = (float)tq[base], im = (float)tq[base + 1];
  float c = cs[s * 64 + j], ss = sn[s * 64 + j];
  tq[base]     = (bf16_t)(r * c - im * ss);
  tq[base + 1] = (bf16_t)(r * ss + im * c);
}

// ---------------------------------------------------------------------------
// 5) Adapter k/v projection (plain wk/wv, no LoRA, no RoPE), padded to 16 keys
//    out layout [B, H, 16, HD] bf16 (rows >= AL zero)
// ---------------------------------------------------------------------------
__global__ void adapter_proj_kernel(const float* __restrict__ adp,
                                    const float* __restrict__ wk,
                                    const float* __restrict__ wv,
                                    bf16_t* __restrict__ ak,
                                    bf16_t* __restrict__ av) {
  int idx = blockIdx.x * 256 + threadIdx.x;   // B*H*16*HD = 262144
  int d = idx & 127;
  int kk = (idx >> 7) & 15;
  int h = (idx >> 11) & 31;
  int b = idx >> 16;
  int o = h * HD + d;
  float sk = 0.f, sv = 0.f;
  if (kk < AL) {
    const float* a  = adp + ((long)b * AL + kk) * DIM;
    const float* rk = wk + (long)o * DIM;
    const float* rv = wv + (long)o * DIM;
    #pragma unroll 4
    for (int c = 0; c < DIM; ++c) { sk += a[c] * rk[c]; sv += a[c] * rv[c]; }
  }
  ak[(long)idx] = (bf16_t)sk;
  av[(long)idx] = (bf16_t)sv;
}

// ---------------------------------------------------------------------------
// 6) Flash attention (causal) + fused gated adapter attention.
//    Grid (S/64, H, B); 128 threads = 4 waves; wave w owns q rows [16w,16w+16).
//    K tile async-copied, V tile staged transposed; double-buffered.
// ---------------------------------------------------------------------------
__global__ __launch_bounds__(128)
void flash_attn_kernel(const bf16_t* __restrict__ Q, const bf16_t* __restrict__ Kb,
                       const bf16_t* __restrict__ Vb, const bf16_t* __restrict__ AK,
                       const bf16_t* __restrict__ AV, const float* __restrict__ gate,
                       bf16_t* __restrict__ O) {
  __shared__ __align__(128) bf16_t Kt[2][32 * 128];    // [k][d]
  __shared__ __align__(128) bf16_t VTt[2][128 * 32];   // [d][k] transposed
  __shared__ __align__(128) bf16_t AVt[128 * 32];      // [d][k], k>=16 zero
  __shared__ __align__(128) bf16_t Pb[4][16 * 32];     // per-wave P staging

  const int b = blockIdx.z, h = blockIdx.y, qb = blockIdx.x * 64;
  const int t = threadIdx.x, lane = t & 31, w = t >> 5;
  const int col = lane & 15, half = lane >> 4;
  const unsigned long long kB = (unsigned long long)Kb;

  auto stage_kv = [&](int buf, int k0) {
    #pragma unroll
    for (int itc = 0; itc < 4; ++itc) {          // K tile: 512 x 16B chunks
      int c = t + itc * 128;
      int r = c >> 4, seg = (c & 15) * 8;
      async_b128(kB, (int)((((long)b * SS + k0 + r) * NH + h) * HD + seg) * 2,
                 &Kt[buf][r * 128 + seg]);
    }
    #pragma unroll
    for (int itc = 0; itc < 4; ++itc) {          // V tile transposed [d][k]
      int u = t + itc * 128;
      int r = u >> 4, dc = (u & 15) * 8;
      bf16x8 vv =
          *(const bf16x8*)(Vb + ((((long)b * SS + k0 + r) * NH + h) * HD + dc));
      #pragma unroll
      for (int j = 0; j < 8; ++j) VTt[buf][(dc + j) * 32 + r] = vv[j];
    }
  };

  // preload Q fragments for this wave's 16 rows (stays in registers)
  bf16x16 qf[4];
  {
    const bf16_t* qrow = Q + (((long)b * SS + qb + w * 16) * NH + h) * HD;
    #pragma unroll
    for (int kf = 0; kf < 4; ++kf) qf[kf] = frag_a(qrow + kf * 32, DIM, lane);
  }

  // stage transposed adapter V (zero padded cols 16..31)
  #pragma unroll
  for (int itc = 0; itc < 32; ++itc) AVt[t + itc * 128] = (bf16_t)0.f;
  __syncthreads();
  #pragma unroll
  for (int itc = 0; itc < 2; ++itc) {
    int u = t + itc * 128;
    int k = u >> 4, dc = (u & 15) * 8;
    bf16x8 vv = *(const bf16x8*)(AV + (((long)b * NH + h) * 16 + k) * HD + dc);
    #pragma unroll
    for (int j = 0; j < 8; ++j) AVt[(dc + j) * 32 + k] = vv[j];
  }

  float m[8], l[8];
  floatx8 acc[8];
  #pragma unroll
  for (int i = 0; i < 8; ++i) { m[i] = -1e30f; l[i] = 0.f; }
  #pragma unroll
  for (int f = 0; f < 8; ++f) acc[f] = (floatx8)(0.f);

  const int kend = qb + 64;        // causal bound for this q-tile
  const int nIter = kend >> 5;
  stage_kv(0, 0);
  for (int itk = 0; itk < nIter; ++itk) {
    const int k0 = itk << 5, cur = itk & 1;
    wait_async();
    wait_ds();                     // V-transpose ds_stores of tile `itk` done
    __syncthreads();
    if (itk + 1 < nIter) stage_kv(cur ^ 1, k0 + 32);   // overlap next copy

    // scores 16q x 32k via WMMA (contraction over HD)
    floatx8 s0 = (floatx8)(0.f), s1 = (floatx8)(0.f);
    #pragma unroll
    for (int kf = 0; kf < 4; ++kf) {
      s0 = wmma_bf16(qf[kf], frag_b(&Kt[cur][0 * 128 + kf * 32], 128, lane), s0);
      s1 = wmma_bf16(qf[kf], frag_b(&Kt[cur][16 * 128 + kf * 32], 128, lane), s1);
    }
    // online softmax (per q row); C rows: half*8 + i
    float corr[8];
    #pragma unroll
    for (int i = 0; i < 8; ++i) {
      int row = qb + w * 16 + half * 8 + i;
      float r0 = s0[i] * SCALE + ((k0 + col      <= row) ? 0.f : -1e9f);
      float r1 = s1[i] * SCALE + ((k0 + 16 + col <= row) ? 0.f : -1e9f);
      float mx = fmaxf(r0, r1);
      #pragma unroll
      for (int d2 = 1; d2 < 16; d2 <<= 1) mx = fmaxf(mx, __shfl_xor(mx, d2, 32));
      float mn = fmaxf(m[i], mx);
      float cf = __builtin_exp2f((m[i] - mn) * LOG2E);
      float p0 = __builtin_exp2f((r0 - mn) * LOG2E);
      float p1 = __builtin_exp2f((r1 - mn) * LOG2E);
      float sm = p0 + p1;
      #pragma unroll
      for (int d2 = 1; d2 < 16; d2 <<= 1) sm += __shfl_xor(sm, d2, 32);
      l[i] = l[i] * cf + sm;
      m[i] = mn;
      corr[i] = cf;
      bf16_t* pr = &Pb[w][(half * 8 + i) * 32];
      pr[col] = (bf16_t)p0;
      pr[col + 16] = (bf16_t)p1;
    }
    #pragma unroll
    for (int f = 0; f < 8; ++f)
      #pragma unroll
      for (int i = 0; i < 8; ++i) acc[f][i] *= corr[i];
    wait_ds();                     // wave-local P tile visible
    bf16x16 pf = frag_a(&Pb[w][0], 32, lane);
    #pragma unroll
    for (int f = 0; f < 8; ++f)
      acc[f] = wmma_bf16(pf, frag_b(&VTt[cur][f * 16 * 32], 32, lane), acc[f]);
  }

  // ---- adapter attention (16 padded keys, separate softmax, tanh gate) ----
  floatx8 as0 = (floatx8)(0.f);
  const bf16_t* akp = AK + ((long)b * NH + h) * 16 * HD;
  #pragma unroll
  for (int kf = 0; kf < 4; ++kf)
    as0 = wmma_bf16(qf[kf], frag_b(akp + kf * 32, HD, lane), as0);
  float l2s[8];
  #pragma unroll
  for (int i = 0; i < 8; ++i) {
    float r0 = as0[i] * SCALE + ((col < AL) ? 0.f : -1e9f);
    float mx = r0;
    #pragma unroll
    for (int d2 = 1; d2 < 16; d2 <<= 1) mx = fmaxf(mx, __shfl_xor(mx, d2, 32));
    float p0 = __builtin_exp2f((r0 - mx) * LOG2E);
    float sm = p0;
    #pragma unroll
    for (int d2 = 1; d2 < 16; d2 <<= 1) sm += __shfl_xor(sm, d2, 32);
    l2s[i] = sm;
    bf16_t* pr = &Pb[w][(half * 8 + i) * 32];
    pr[col] = (bf16_t)p0;
    pr[col + 16] = (bf16_t)0.f;
  }
  wait_ds();
  bf16x16 pf2 = frag_a(&Pb[w][0], 32, lane);
  floatx8 acc2[8];
  #pragma unroll
  for (int f = 0; f < 8; ++f) {
    acc2[f] = (floatx8)(0.f);
    acc2[f] = wmma_bf16(pf2, frag_b(&AVt[f * 16 * 32], 32, lane), acc2[f]);
  }
  const float g = tanhf(gate[h]);
  #pragma unroll
  for (int f = 0; f < 8; ++f)
    #pragma unroll
    for (int i = 0; i < 8; ++i) {
      int row = qb + w * 16 + half * 8 + i;
      float o = acc[f][i] / l[i] + g * acc2[f][i] / l2s[i];
      O[(((long)b * SS + row) * DIM) + h * HD + f * 16 + col] = (bf16_t)o;
    }
}

// ---------------------------------------------------------------------------
// Host launcher
// ---------------------------------------------------------------------------
extern "C" void kernel_launch(void* const* d_in, const int* in_sizes, int n_in,
                              void* d_out, int out_size, void* d_ws, size_t ws_size,
                              hipStream_t stream) {
  (void)in_sizes; (void)n_in; (void)out_size; (void)ws_size;
  const float* x    = (const float*)d_in[0];
  const float* adp  = (const float*)d_in[1];
  const float* wq   = (const float*)d_in[2];
  const float* wk   = (const float*)d_in[3];
  const float* wv   = (const float*)d_in[4];
  const float* wo   = (const float*)d_in[5];
  const float* lq1  = (const float*)d_in[6];
  const float* lq2  = (const float*)d_in[7];
  const float* lk1  = (const float*)d_in[8];
  const float* lk2  = (const float*)d_in[9];
  const float* lv1  = (const float*)d_in[10];
  const float* lv2  = (const float*)d_in[11];
  const float* lo1  = (const float*)d_in[12];
  const float* lo2  = (const float*)d_in[13];
  const float* gate = (const float*)d_in[14];
  const float* fcos = (const float*)d_in[15];
  const float* fsin = (const float*)d_in[16];
  // d_in[17] = mask (causal, recomputed on device), d_in[18] = start_pos (0)

  const long WEL = (long)DIM * DIM;        // 16,777,216 elements
  bf16_t* base   = (bf16_t*)d_ws;
  bf16_t* weffq  = base + 0 * WEL;
  bf16_t* weffk  = base + 1 * WEL;
  bf16_t* weffv  = base + 2 * WEL;
  bf16_t* weffo  = base + 3 * WEL;
  bf16_t* xbf    = base + 4 * WEL;         // [M, DIM]
  bf16_t* qbf    = base + 5 * WEL;         // [B,S,H,HD]
  bf16_t* kbf    = base + 6 * WEL;
  bf16_t* vbf    = base + 7 * WEL;
  bf16_t* attnbf = base + 8 * WEL;         // [B,S,DIM]
  bf16_t* akbf   = base + 9 * WEL;         // [B,H,16,HD]
  bf16_t* avbf   = akbf + (long)BB * NH * 16 * HD;

  const int NELX = (int)(MROWS * DIM);     // 16,777,216

  // 1) x -> bf16
  cvt_bf16_kernel<<<NELX / 256, 256, 0, stream>>>(x, xbf, NELX);
  // 2) fold LoRA into effective weights
  merge_weight_kernel<<<NELX / 256, 256, 0, stream>>>(wq, lq2, lq1, weffq);
  merge_weight_kernel<<<NELX / 256, 256, 0, stream>>>(wk, lk2, lk1, weffk);
  merge_weight_kernel<<<NELX / 256, 256, 0, stream>>>(wv, lv2, lv1, weffv);
  merge_weight_kernel<<<NELX / 256, 256, 0, stream>>>(wo, lo2, lo1, weffo);
  // 3) QKV projections (bf16 WMMA GEMMs)
  dim3 gg(DIM / 128, (int)(MROWS / 128));
  gemm_bf16_kernel<<<gg, 256, 0, stream>>>(xbf, weffq, qbf, (int)MROWS, DIM, DIM, 0);
  gemm_bf16_kernel<<<gg, 256, 0, stream>>>(xbf, weffk, kbf, (int)MROWS, DIM, DIM, 0);
  gemm_bf16_kernel<<<gg, 256, 0, stream>>>(xbf, weffv, vbf, (int)MROWS, DIM, DIM, 0);
  // 4) RoPE on q, k
  const int NROPE = BB * SS * NH * (HD / 2);   // 8,388,608
  rope_kernel<<<NROPE / 256, 256, 0, stream>>>(qbf, fcos, fsin);
  rope_kernel<<<NROPE / 256, 256, 0, stream>>>(kbf, fcos, fsin);
  // 5) adapter k/v (plain weights, padded to 16 keys)
  adapter_proj_kernel<<<(BB * NH * 16 * HD) / 256, 256, 0, stream>>>(
      adp, wk, wv, akbf, avbf);
  // 6) fused flash + adapter attention
  flash_attn_kernel<<<dim3(SS / 64, NH, BB), 128, 0, stream>>>(
      qbf, kbf, vbf, akbf, avbf, gate, attnbf);
  // 7) output projection (fp32 out)
  gemm_bf16_kernel<<<gg, 256, 0, stream>>>(attnbf, weffo, (float*)d_out,
                                           (int)MROWS, DIM, DIM, 1);
}
